// ResidentTransformer_63883343561204
// MI455X (gfx1250) — matrix operand order repaired and checked by reading, not compiled, not run on hardware
//
#include <hip/hip_runtime.h>
#include <cmath>

typedef _Float16 f16;
typedef _Float16 v16h __attribute__((ext_vector_type(16)));
typedef _Float16 v8h  __attribute__((ext_vector_type(8)));
typedef float    v8f  __attribute__((ext_vector_type(8)));

#define NL  8
#define NB  4
#define HDIM 2048
#define NHEAD 16
#define HD  128
#define SEQ 2048          // kv rows 0..SEQ, SEQ is the fresh row
#define H3  (3*HDIM)      // 6144
#define FF  (4*HDIM)      // 8192
#define LNEPS 1e-5f

// ---------------------------------------------------------------- positional
__global__ void pos_add_kernel(const f16* __restrict__ x_in,
                               const int* __restrict__ cur_pos,
                               f16* __restrict__ x) {
  int idx = blockIdx.x * blockDim.x + threadIdx.x;          // NB*HDIM
  if (idx >= NB * HDIM) return;
  int j = idx % HDIM;
  float pos = (float)cur_pos[0];                            // q_len==1 -> pos+0
  float div = expf((float)((j >> 1) << 1) * (-9.210340371976184f / (float)HDIM));
  float ang = pos * div;
  float pe  = (j & 1) ? cosf(ang) : sinf(ang);
  x[idx] = (f16)((float)x_in[idx] + pe);
}

// ---------------------------------------------------------------- layernorm
__global__ void ln_kernel(const f16* __restrict__ x,
                          const f16* __restrict__ w,
                          const f16* __restrict__ b,
                          f16* __restrict__ y) {
  __shared__ float ls[256], ls2[256];
  int bb = blockIdx.x, t = threadIdx.x;
  const f16* row = x + (size_t)bb * HDIM;
  float s = 0.f, s2 = 0.f;
  for (int j = t; j < HDIM; j += 256) { float v = (float)row[j]; s += v; s2 += v * v; }
  ls[t] = s; ls2[t] = s2; __syncthreads();
  for (int off = 128; off > 0; off >>= 1) {
    if (t < off) { ls[t] += ls[t + off]; ls2[t] += ls2[t + off]; }
    __syncthreads();
  }
  float mean = ls[0] * (1.0f / HDIM);
  float var  = ls2[0] * (1.0f / HDIM) - mean * mean;
  float rstd = rsqrtf(var + LNEPS);
  for (int j = t; j < HDIM; j += 256) {
    float v = (float)row[j];
    y[(size_t)bb * HDIM + j] = (f16)(((v - mean) * rstd) * (float)w[j] + (float)b[j]);
  }
}

// --------------------------------------------------- WMMA GEMV (x @ W^T)
// C[batch<4][n] = sum_k xn[batch][k] * W[n][k], via D = A(16xK Wtile) x B(Kx16 x^T)
// Invalid B columns are CLAMPED, not zeroed: D columns are independent, and we
// only read columns 0..3, so no EXEC divergence in the streaming loop.
// act: 0=none 1=exact gelu ; resid optional ; writes f16 (out_h) or f32 (out_f)
__global__ void wmma_gemv_kernel(const f16* __restrict__ xn,
                                 const f16* __restrict__ W,
                                 int K, int N, int act,
                                 const f16* __restrict__ resid,
                                 f16* __restrict__ out_h,
                                 float* __restrict__ out_f) {
  int wave = threadIdx.x >> 5;
  int lane = threadIdx.x & 31;
  int n0   = blockIdx.x * 128 + wave * 16;
  int col  = lane & 15;            // A: row M ; B/C/D: column N (=batch)
  int hi   = lane >> 4;
  int abase = hi ? 16 : 0;         // A layout: lanes 0-15 K=0..15, 16-31 K=16..31
  int kb    = hi ? 16 : 0;         // B layout: lanes 0-15 K=0..15, 16-31 K=16..31
  const f16* wrow = W + (size_t)(n0 + col) * K;
  const f16* xrow = xn + (size_t)(col & (NB - 1)) * K;   // clamp: cols >=4 unread

  v8f acc = {};
  for (int k0 = 0; k0 < K; k0 += 32) {
    v8h alo = *(const v8h*)(wrow + k0 + abase);
    v8h ahi = *(const v8h*)(wrow + k0 + 8 + abase);
    v16h a  = __builtin_shufflevector(alo, ahi, 0,1,2,3,4,5,6,7,8,9,10,11,12,13,14,15);
    v16h bf = *(const v16h*)(xrow + k0 + kb);
    __builtin_prefetch(wrow + k0 + 512, 0, 1);   // global_prefetch_b8: stream weights
    acc = __builtin_amdgcn_wmma_f32_16x16x32_f16(false, a, false, bf,
                                                 (short)0, acc, false, false);
  }
  if (col < NB) {
    for (int r = 0; r < 8; ++r) {
      int m = n0 + r + 8 * hi;
      float v = acc[r];
      if (act == 1) v = 0.5f * v * (1.0f + erff(v * 0.70710678118654752f));
      if (resid) v += (float)resid[(size_t)col * N + m];
      if (out_f) out_f[(size_t)col * N + m] = v;
      else       out_h[(size_t)col * N + m] = (f16)v;
    }
  }
}

// ------------------------------------------------------------- attention
// one block per (b, head): WMMA q.K scores -> softmax -> attn.V -> residual
__global__ void attn_kernel(const f16* __restrict__ qkv,   // [NB][H3]
                            const f16* __restrict__ kc,    // layer base [NB][NH][SEQ+1][HD]
                            const f16* __restrict__ vc,
                            f16* __restrict__ x) {         // [NB][HDIM] in/out
  __shared__ float sc[SEQ + 16];
  __shared__ float red[256];

  int bh = blockIdx.x;
  int b = bh / NHEAD, h = bh % NHEAD;
  const f16* q    = qkv + (size_t)b * H3 + (size_t)h * (3 * HD);
  const f16* knew = q + HD;
  const f16* vnew = q + 2 * HD;
  const f16* kbas = kc + (size_t)(b * NHEAD + h) * (SEQ + 1) * HD;
  const f16* vbas = vc + (size_t)(b * NHEAD + h) * (SEQ + 1) * HD;

  int t = threadIdx.x, wave = t >> 5, lane = t & 31;
  int col = lane & 15, hi = lane >> 4;
  int abase = hi ? 16 : 0, kb = hi ? 16 : 0;

  // ---- scores: 129 tiles of 16 keys across 8 waves.
  // All lanes load q for B (only D column 0 is read); rows past SEQ read the
  // (valid) fresh-k row and their scores are simply not written back.
  for (int tile = wave; tile < (SEQ + 16) / 16; tile += 8) {
    int s0 = tile * 16;
    int srow = s0 + col;
    const f16* arow = (srow < SEQ) ? (kbas + (size_t)srow * HD) : knew;
    v8f acc = {};
    for (int k0 = 0; k0 < HD; k0 += 32) {
      v8h alo = *(const v8h*)(arow + k0 + abase);
      v8h ahi = *(const v8h*)(arow + k0 + 8 + abase);
      v16h a  = __builtin_shufflevector(alo, ahi, 0,1,2,3,4,5,6,7,8,9,10,11,12,13,14,15);
      v16h bf = *(const v16h*)(q + k0 + kb);      // same q in every column
      acc = __builtin_amdgcn_wmma_f32_16x16x32_f16(false, a, false, bf,
                                                   (short)0, acc, false, false);
    }
    if (col == 0) {                               // lanes 0 & 16 own column 0
      for (int r = 0; r < 8; ++r) {
        int s = s0 + r + 8 * hi;
        if (s <= SEQ) sc[s] = acc[r] * 0.08838834764831845f;  // 1/sqrt(128)
      }
    }
  }
  __syncthreads();

  // ---- softmax over SEQ+1
  float m = -1e30f;
  for (int s = t; s <= SEQ; s += 256) m = fmaxf(m, sc[s]);
  red[t] = m; __syncthreads();
  for (int off = 128; off > 0; off >>= 1) {
    if (t < off) red[t] = fmaxf(red[t], red[t + off]);
    __syncthreads();
  }
  float mx = red[0]; __syncthreads();
  float sum = 0.f;
  for (int s = t; s <= SEQ; s += 256) { float e = __expf(sc[s] - mx); sc[s] = e; sum += e; }
  red[t] = sum; __syncthreads();
  for (int off = 128; off > 0; off >>= 1) {
    if (t < off) red[t] += red[t + off];
    __syncthreads();
  }
  float inv = 1.0f / red[0];
  __syncthreads();

  // ---- out = attn . V  (bandwidth bound: 128 lanes own d, 2-way split on s)
  int d = t & (HD - 1);
  int half = t >> 7;
  float o = 0.f;
  for (int s = half; s < SEQ; s += 2)
    o += sc[s] * (float)vbas[(size_t)s * HD + d];
  if ((SEQ & 1) == half)                           // fresh row s == SEQ
    o += sc[SEQ] * (float)vnew[d];
  red[t] = o; __syncthreads();
  if (half == 0) {
    float tot = (red[t] + red[t + 128]) * inv;
    size_t xi = (size_t)b * HDIM + (size_t)h * HD + d;
    x[xi] = (f16)((float)x[xi] + tot);             // residual add
  }
}

// --------------------------------------------------------------- softmax out
__global__ void softmax_kernel(const float* __restrict__ logits, f16* __restrict__ out) {
  __shared__ float red[256];
  int b = blockIdx.x, t = threadIdx.x;
  const float* row = logits + (size_t)b * HDIM;
  float m = -1e30f;
  for (int j = t; j < HDIM; j += 256) m = fmaxf(m, row[j]);
  red[t] = m; __syncthreads();
  for (int off = 128; off > 0; off >>= 1) {
    if (t < off) red[t] = fmaxf(red[t], red[t + off]);
    __syncthreads();
  }
  float mx = red[0]; __syncthreads();
  float s = 0.f;
  for (int j = t; j < HDIM; j += 256) s += __expf(row[j] - mx);
  red[t] = s; __syncthreads();
  for (int off = 128; off > 0; off >>= 1) {
    if (t < off) red[t] += red[t + off];
    __syncthreads();
  }
  float inv = 1.0f / red[0];
  for (int j = t; j < HDIM; j += 256)
    out[(size_t)b * HDIM + j] = (f16)(__expf(row[j] - mx) * inv);
}

// ---------------------------------------------------------------- launcher
extern "C" void kernel_launch(void* const* d_in, const int* in_sizes, int n_in,
                              void* d_out, int out_size, void* d_ws, size_t ws_size,
                              hipStream_t stream) {
  const f16* x_in    = (const f16*)d_in[0];
  const f16* qkv_w   = (const f16*)d_in[1];
  const f16* k_cache = (const f16*)d_in[2];
  const f16* v_cache = (const f16*)d_in[3];
  const f16* ffn1_w  = (const f16*)d_in[4];
  const f16* ffn2_w  = (const f16*)d_in[5];
  const f16* out_w   = (const f16*)d_in[6];
  const f16* ln_w    = (const f16*)d_in[7];
  const f16* ln_b    = (const f16*)d_in[8];
  const int* cur     = (const int*)d_in[9];

  char* ws = (char*)d_ws;
  f16*   x      = (f16*)ws;   ws += (size_t)NB * HDIM * 2;   // residual stream
  f16*   xn     = (f16*)ws;   ws += (size_t)NB * HDIM * 2;   // LN output
  f16*   qkv    = (f16*)ws;   ws += (size_t)NB * H3   * 2;   // qkv projection
  f16*   mid    = (f16*)ws;   ws += (size_t)NB * FF   * 2;   // ffn hidden
  float* logits = (float*)ws;                                 // final logits

  const size_t KVL = (size_t)NB * NHEAD * (SEQ + 1) * HD;     // per-layer kv elems

  pos_add_kernel<<<(NB * HDIM + 255) / 256, 256, 0, stream>>>(x_in, cur, x);

  for (int l = 0; l < NL; ++l) {
    ln_kernel<<<NB, 256, 0, stream>>>(x, ln_w, ln_b, xn);
    wmma_gemv_kernel<<<H3 / 128, 256, 0, stream>>>(
        xn, qkv_w + (size_t)l * H3 * HDIM, HDIM, H3, 0, nullptr, qkv, nullptr);
    attn_kernel<<<NB * NHEAD, 256, 0, stream>>>(
        qkv, k_cache + (size_t)l * KVL, v_cache + (size_t)l * KVL, x);
    ln_kernel<<<NB, 256, 0, stream>>>(x, ln_w, ln_b, xn);
    wmma_gemv_kernel<<<FF / 128, 256, 0, stream>>>(
        xn, ffn1_w + (size_t)l * FF * HDIM, HDIM, FF, 1, nullptr, mid, nullptr);
    wmma_gemv_kernel<<<HDIM / 128, 256, 0, stream>>>(
        mid, ffn2_w + (size_t)l * HDIM * FF, FF, HDIM, 0, x, x, nullptr);
  }

  ln_kernel<<<NB, 256, 0, stream>>>(x, ln_w, ln_b, xn);
  wmma_gemv_kernel<<<HDIM / 128, 256, 0, stream>>>(
      xn, out_w, HDIM, HDIM, 0, nullptr, nullptr, logits);
  softmax_kernel<<<NB, 256, 0, stream>>>(logits, (f16*)d_out);
}
